// MultiHeadDifferentialAttention_51891794870797
// MI455X (gfx1250) — matrix operand order repaired and checked by reading, not compiled, not run on hardware
//
#include <hip/hip_runtime.h>
#include <hip/hip_bf16.h>

// ---------------------------------------------------------------------------
// Multi-head differential attention, MI455X (gfx1250), bf16 WMMA pipeline.
// B=2, T=2048, C=1024, H=16, D=64.
// GEMMs: 256x64 block tile, 32 rows/wave, K-tile 64; all LDS fragment traffic
// is ds_load_b128 (contiguous per-lane layouts, B operands staged N-major).
// ---------------------------------------------------------------------------

constexpr int Bc = 2;
constexpr int Tc = 2048;
constexpr int Cc = 1024;
constexpr int Hc = 16;
constexpr int Dc = 64;
constexpr int BT = Bc * Tc;                               // 4096
constexpr long long BHTD = (long long)Bc * Hc * Tc * Dc;  // 4,194,304

typedef __attribute__((ext_vector_type(16))) unsigned short v16u;
typedef __attribute__((ext_vector_type(16))) __bf16         v16bf;
typedef __attribute__((ext_vector_type(8)))  float          v8f;

// ---- helpers --------------------------------------------------------------

static __device__ __forceinline__ unsigned short f32_to_bf16(float f) {
  union { float f; unsigned u; } v; v.f = f;
  unsigned r = v.u + 0x7FFFu + ((v.u >> 16) & 1u);   // round-to-nearest-even
  return (unsigned short)(r >> 16);
}

static __device__ __forceinline__ v16u ld2x16(const unsigned short* p0,
                                              const unsigned short* p1) {
  union { v16u v; uint4 q[2]; } u;
  u.q[0] = *reinterpret_cast<const uint4*>(p0);
  u.q[1] = *reinterpret_cast<const uint4*>(p1);
  return u.v;
}

// A fragment (16x32, M x K), source row-major (M x K), ld in elements.
// Lane L<16: row M=L,   e[0..7]=K 0..7,   e[8..15]=K 16..23
// Lane L>=16: row M=L-16, e[0..7]=K 8..15, e[8..15]=K 24..31
// Requires ld*2 % 16 == 0 -> 2x ds_load_b128 per fragment.
static __device__ __forceinline__ v16u load_a_frag(const unsigned short* s, int ld) {
  int lane = threadIdx.x & 31;
  const unsigned short* base = s + (lane & 15) * ld + ((lane >> 4) << 3);
  return ld2x16(base, base + 16);
}

// B fragment (32x16, K x N), source stored N-major (N x K row-major), ld in elts.
// Lane L<16: col N=L, e[0..7]=K 0..7, e[8..15]=K 16..23.  Lane L>=16: col N=L-16, e[0..7]=K 8..15, e[8..15]=K 24..31.
// Two 8-element runs per lane -> 2x ds_load_b128.
static __device__ __forceinline__ v16u load_bt_frag(const unsigned short* s, int ld) {
  int lane = threadIdx.x & 31;
  const unsigned short* base = s + (lane & 15) * ld + ((lane >> 4) << 3);
  return ld2x16(base, base + 16);
}

static __device__ __forceinline__ v8f wmma_bf16(v16u a, v16u b, v8f c) {
  return __builtin_amdgcn_wmma_f32_16x16x32_bf16(
      false, __builtin_bit_cast(v16bf, a),
      false, __builtin_bit_cast(v16bf, b),
      (short)0, c, false, false);
}

// ---- stage 0: fp32 -> bf16 conversion ------------------------------------

__global__ void cvt_bf16_kernel(const float* __restrict__ in,
                                unsigned short* __restrict__ out, int n) {
  int i = blockIdx.x * blockDim.x + threadIdx.x;
  if (i < n) out[i] = f32_to_bf16(in[i]);
}

__global__ void lam_kernel(const float* __restrict__ l, float* __restrict__ lam) {
  if (threadIdx.x == 0 && blockIdx.x == 0)
    *lam = __expf(l[0] * l[1]) - __expf(l[2] * l[3]) + 0.2f;   // INITIAL_LAMBDA = 0.2
}

// ---- stage 1: per-head projections ----------------------------------------
// For p in {q1,k1,q2,k2,v}, h in heads:  (T x C) @ (C x D) -> (T x D) bf16.
// grid = (T/256, 5*B*H), block = 256 = 8 waves; wave w owns rows 32w..32w+31.
// K-tile 64: 16 WMMA per wave per barrier pair, 24 b128 LDS loads per 16 WMMA.

constexpr int LKT = 72;  // K-tile 64 + 8 pad (144 B/row, 16B aligned)

__global__ __launch_bounds__(256)
void proj_gemm_kernel(const unsigned short* __restrict__ xb,   // (B*T, C)
                      const unsigned short* __restrict__ w5,   // (5,H,C,D)
                      unsigned short* __restrict__ qkv)        // (5,B,H,T,D)
{
  __shared__ __align__(16) unsigned short As[256 * LKT];  // 256 rows x 64 k
  __shared__ __align__(16) unsigned short Wt[64 * LKT];   // N-major: 64 n x 64 k

  int mtile = blockIdx.x;
  int z     = blockIdx.y;
  int p  = z / (Bc * Hc);
  int bh = z % (Bc * Hc);
  int b  = bh / Hc, h = bh % Hc;

  const unsigned short* A = xb + ((size_t)b * Tc + (size_t)mtile * 256) * Cc;
  const unsigned short* W = w5 + ((size_t)p * Hc + h) * (size_t)Cc * Dc;
  unsigned short* O = qkv + ((((size_t)p * Bc + b) * Hc + h) * Tc + (size_t)mtile * 256) * Dc;

  int tid = threadIdx.x, lane = tid & 31, wid = tid >> 5;
  v8f acc[2][4] = {};

  for (int kk = 0; kk < Cc; kk += 64) {
    __syncthreads();
    // A tile 256x64, 16B vector copies (8 per thread)
    for (int i = tid; i < 256 * 8; i += 256) {
      int r = i >> 3, c8 = (i & 7) * 8;
      *reinterpret_cast<uint4*>(&As[r * LKT + c8]) =
          *reinterpret_cast<const uint4*>(&A[(size_t)r * Cc + kk + c8]);
    }
    // W tile 64x64 -> transposed into Wt[64 n][64 k]
    for (int i = tid; i < 64 * 8; i += 256) {
      int r = i >> 3, c0 = (i & 7) * 8;
      union { uint4 q; unsigned short u[8]; } t;
      t.q = *reinterpret_cast<const uint4*>(&W[(size_t)(kk + r) * Dc + c0]);
#pragma unroll
      for (int j = 0; j < 8; ++j) Wt[(c0 + j) * LKT + r] = t.u[j];
    }
    __syncthreads();
#pragma unroll
    for (int kh = 0; kh < 64; kh += 32) {
      v16u a0 = load_a_frag(&As[(wid * 32) * LKT + kh], LKT);
      v16u a1 = load_a_frag(&As[(wid * 32 + 16) * LKT + kh], LKT);
#pragma unroll
      for (int n4 = 0; n4 < 4; ++n4) {
        v16u bf = load_bt_frag(&Wt[(n4 * 16) * LKT + kh], LKT);
        acc[0][n4] = wmma_bf16(a0, bf, acc[0][n4]);
        acc[1][n4] = wmma_bf16(a1, bf, acc[1][n4]);
      }
    }
  }

  int col = lane & 15, rbase = (lane >> 4) * 8;
#pragma unroll
  for (int half = 0; half < 2; ++half)
#pragma unroll
    for (int n4 = 0; n4 < 4; ++n4)
#pragma unroll
      for (int r = 0; r < 8; ++r)
        O[(size_t)(wid * 32 + half * 16 + rbase + r) * Dc + n4 * 16 + col] =
            f32_to_bf16(acc[half][n4][r]);
}

// ---- stage 2: differential flash attention --------------------------------
// grid = (T/64, B*H); block = 256 = 8 waves.
// Waves 0-3: stream 1 (q1,k1) rows 0..63 (16 each); waves 4-7: stream 2 (q2,k2).

constexpr int LQ = 72;  // 64 + 8 pad (144 bytes/row, 16B aligned)

__global__ __launch_bounds__(256)
void diff_attn_kernel(const unsigned short* __restrict__ qkv,  // (5,B,H,T,D)
                      const float* __restrict__ lamp,
                      unsigned short* __restrict__ attn)       // (B,T,C) bf16
{
  __shared__ __align__(16) unsigned short Qall[2][64 * LQ];  // q-rows x d (row-major)
  __shared__ __align__(16) unsigned short Kall[2][64 * LQ];  // keys x d (= N-major for S)
  __shared__ __align__(16) unsigned short Vt[64 * LQ];       // d x keys (N-major for PV)
  __shared__ __align__(16) unsigned short Pst[8 * 16 * LQ];  // per-wave P staging
  float* Comb = (float*)Pst;                                 // reuse for combine (16 KB)

  int qt = blockIdx.x;
  int bh = blockIdx.y;
  int b = bh >> 4, h = bh & 15;

  size_t base = ((size_t)b * Hc + h) * (size_t)Tc * Dc;
  const unsigned short* Q1 = qkv + 0 * BHTD + base + (size_t)qt * 64 * Dc;
  const unsigned short* K1 = qkv + 1 * BHTD + base;
  const unsigned short* Q2 = qkv + 2 * BHTD + base + (size_t)qt * 64 * Dc;
  const unsigned short* K2 = qkv + 3 * BHTD + base;
  const unsigned short* Vp = qkv + 4 * BHTD + base;

  int tid = threadIdx.x, lane = tid & 31, wid = tid >> 5;
  int strm = wid >> 2;
  int wrow = (wid & 3) * 16;
  int col = lane & 15, rbase = (lane >> 4) * 8;
  int q0 = qt * 64;

  // Q tiles once (both streams), 16B vector copies
  for (int i = tid; i < 64 * 8; i += 256) {
    int r = i >> 3, c8 = (i & 7) * 8;
    *reinterpret_cast<uint4*>(&Qall[0][r * LQ + c8]) =
        *reinterpret_cast<const uint4*>(&Q1[(size_t)r * Dc + c8]);
    *reinterpret_cast<uint4*>(&Qall[1][r * LQ + c8]) =
        *reinterpret_cast<const uint4*>(&Q2[(size_t)r * Dc + c8]);
  }

  const unsigned short* Qs = &Qall[strm][0];
  const unsigned short* Ks = &Kall[strm][0];

  v8f o[4] = {};
  float m[8], l[8];
#pragma unroll
  for (int r = 0; r < 8; ++r) { m[r] = -1e30f; l[r] = 0.0f; }

  const float sscale = 0.125f;  // 1/sqrt(64)

  for (int kb = 0; kb <= qt; ++kb) {
    int k0 = kb * 64;
    __syncthreads();
    for (int i = tid; i < 64 * 8; i += 256) {
      int r = i >> 3, c8 = (i & 7) * 8;
      *reinterpret_cast<uint4*>(&Kall[0][r * LQ + c8]) =
          *reinterpret_cast<const uint4*>(&K1[(size_t)(k0 + r) * Dc + c8]);
      *reinterpret_cast<uint4*>(&Kall[1][r * LQ + c8]) =
          *reinterpret_cast<const uint4*>(&K2[(size_t)(k0 + r) * Dc + c8]);
      union { uint4 q; unsigned short u[8]; } t;   // V transposed: Vt[d][key]
      t.q = *reinterpret_cast<const uint4*>(&Vp[(size_t)(k0 + r) * Dc + c8]);
#pragma unroll
      for (int j = 0; j < 8; ++j) Vt[(c8 + j) * LQ + r] = t.u[j];
    }
    __syncthreads();

    // S (16x64) = Q_strip (16x64) @ K_tileᵀ ; K tile is keys x d == N-major
    v8f sa[4] = {};
#pragma unroll
    for (int dh = 0; dh < 64; dh += 32) {
      v16u a = load_a_frag(&Qs[wrow * LQ + dh], LQ);
#pragma unroll
      for (int n4 = 0; n4 < 4; ++n4) {
        v16u bf = load_bt_frag(&Ks[(n4 * 16) * LQ + dh], LQ);
        sa[n4] = wmma_bf16(a, bf, sa[n4]);
      }
    }

    // causal mask + scale + online softmax
    float p[4][8];
#pragma unroll
    for (int r = 0; r < 8; ++r) {
      int qi = q0 + wrow + rbase + r;
      float mx = -1e30f;
#pragma unroll
      for (int n4 = 0; n4 < 4; ++n4) {
        int ki = k0 + n4 * 16 + col;
        float v = (ki <= qi) ? sa[n4][r] * sscale : -1e30f;
        p[n4][r] = v;
        mx = fmaxf(mx, v);
      }
      mx = fmaxf(mx, __shfl_xor(mx, 1, 32));
      mx = fmaxf(mx, __shfl_xor(mx, 2, 32));
      mx = fmaxf(mx, __shfl_xor(mx, 4, 32));
      mx = fmaxf(mx, __shfl_xor(mx, 8, 32));
      float mnew = fmaxf(m[r], mx);
      float alpha = __expf(m[r] - mnew);
      float rs = 0.0f;
#pragma unroll
      for (int n4 = 0; n4 < 4; ++n4) {
        float e = __expf(p[n4][r] - mnew);
        p[n4][r] = e;
        rs += e;
      }
      rs += __shfl_xor(rs, 1, 32);
      rs += __shfl_xor(rs, 2, 32);
      rs += __shfl_xor(rs, 4, 32);
      rs += __shfl_xor(rs, 8, 32);
      l[r] = l[r] * alpha + rs;
      m[r] = mnew;
#pragma unroll
      for (int n4 = 0; n4 < 4; ++n4) o[n4][r] *= alpha;
    }

    // stage P (C-layout -> bf16 LDS row-major) for the P @ V WMMA
    unsigned short* Pw = &Pst[wid * 16 * LQ];
#pragma unroll
    for (int n4 = 0; n4 < 4; ++n4)
#pragma unroll
      for (int r = 0; r < 8; ++r)
        Pw[(rbase + r) * LQ + n4 * 16 + col] = f32_to_bf16(p[n4][r]);
    __syncthreads();

#pragma unroll
    for (int kh = 0; kh < 64; kh += 32) {
      v16u a = load_a_frag(&Pw[kh], LQ);
#pragma unroll
      for (int n4 = 0; n4 < 4; ++n4) {
        v16u bf = load_bt_frag(&Vt[(n4 * 16) * LQ + kh], LQ);
        o[n4] = wmma_bf16(a, bf, o[n4]);
      }
    }
  }

  // combine streams: out = (O1/l1 - lam * O2/l2) * (1 - 0.2)
  float lam = *lamp;
  __syncthreads();
  if (strm == 1) {
#pragma unroll
    for (int n4 = 0; n4 < 4; ++n4)
#pragma unroll
      for (int r = 0; r < 8; ++r)
        Comb[(wrow + rbase + r) * 64 + n4 * 16 + col] = o[n4][r] / l[r];
  }
  __syncthreads();
  if (strm == 0) {
#pragma unroll
    for (int n4 = 0; n4 < 4; ++n4)
#pragma unroll
      for (int r = 0; r < 8; ++r) {
        int row = wrow + rbase + r;
        int d   = n4 * 16 + col;
        float v1 = o[n4][r] / l[r];
        float v2 = Comb[row * 64 + d];
        float res = (v1 - lam * v2) * 0.8f;
        attn[((size_t)b * Tc + q0 + row) * Cc + h * 64 + d] = f32_to_bf16(res);
      }
  }
}

// ---- stage 3: output projection  (4096x1024) @ (1024x1024) + bias ---------
// grid = (BT/256, C/64), block 256; same tiling as proj_gemm.

__global__ __launch_bounds__(256)
void out_gemm_kernel(const unsigned short* __restrict__ Ab,  // (BT, C) bf16
                     const unsigned short* __restrict__ Wb,  // (C, C) bf16
                     const float* __restrict__ bias,
                     float* __restrict__ out)                // (BT, C) f32
{
  __shared__ __align__(16) unsigned short As[256 * LKT];
  __shared__ __align__(16) unsigned short Wt[64 * LKT];

  int mt = blockIdx.x, nt = blockIdx.y;
  const unsigned short* A = Ab + (size_t)mt * 256 * Cc;
  const unsigned short* W = Wb + (size_t)nt * 64;
  float* O = out + (size_t)mt * 256 * Cc + (size_t)nt * 64;

  int tid = threadIdx.x, lane = tid & 31, wid = tid >> 5;
  v8f acc[2][4] = {};

  for (int kk = 0; kk < Cc; kk += 64) {
    __syncthreads();
    for (int i = tid; i < 256 * 8; i += 256) {
      int r = i >> 3, c8 = (i & 7) * 8;
      *reinterpret_cast<uint4*>(&As[r * LKT + c8]) =
          *reinterpret_cast<const uint4*>(&A[(size_t)r * Cc + kk + c8]);
    }
    for (int i = tid; i < 64 * 8; i += 256) {
      int r = i >> 3, c0 = (i & 7) * 8;
      union { uint4 q; unsigned short u[8]; } t;
      t.q = *reinterpret_cast<const uint4*>(&W[(size_t)(kk + r) * Cc + c0]);
#pragma unroll
      for (int j = 0; j < 8; ++j) Wt[(c0 + j) * LKT + r] = t.u[j];
    }
    __syncthreads();
#pragma unroll
    for (int kh = 0; kh < 64; kh += 32) {
      v16u a0 = load_a_frag(&As[(wid * 32) * LKT + kh], LKT);
      v16u a1 = load_a_frag(&As[(wid * 32 + 16) * LKT + kh], LKT);
#pragma unroll
      for (int n4 = 0; n4 < 4; ++n4) {
        v16u bf = load_bt_frag(&Wt[(n4 * 16) * LKT + kh], LKT);
        acc[0][n4] = wmma_bf16(a0, bf, acc[0][n4]);
        acc[1][n4] = wmma_bf16(a1, bf, acc[1][n4]);
      }
    }
  }

  int col = lane & 15, rbase = (lane >> 4) * 8;
#pragma unroll
  for (int half = 0; half < 2; ++half)
#pragma unroll
    for (int n4 = 0; n4 < 4; ++n4)
#pragma unroll
      for (int r = 0; r < 8; ++r) {
        int n = n4 * 16 + col;
        O[(size_t)(wid * 32 + half * 16 + rbase + r) * Cc + n] =
            acc[half][n4][r] + bias[nt * 64 + n];
      }
}

// ---------------------------------------------------------------------------

extern "C" void kernel_launch(void* const* d_in, const int* in_sizes, int n_in,
                              void* d_out, int out_size, void* d_ws, size_t ws_size,
                              hipStream_t stream) {
  (void)in_sizes; (void)n_in; (void)out_size; (void)ws_size;

  const float* x       = (const float*)d_in[0];
  const float* key1    = (const float*)d_in[1];
  const float* key2    = (const float*)d_in[2];
  const float* query1  = (const float*)d_in[3];
  const float* query2  = (const float*)d_in[4];
  const float* value   = (const float*)d_in[5];
  const float* lambdas = (const float*)d_in[6];
  const float* proj_w  = (const float*)d_in[7];
  const float* proj_b  = (const float*)d_in[8];
  float* out = (float*)d_out;

  char* ws = (char*)d_ws;
  size_t off = 0;
  auto take = [&](size_t bytes) -> char* {
    char* p = ws + off;
    off = (off + bytes + 255) & ~(size_t)255;
    return p;
  };

  const size_t wsz = (size_t)Hc * Cc * Dc;                 // 1,048,576 per weight
  unsigned short* xb    = (unsigned short*)take((size_t)BT * Cc * 2);
  unsigned short* w5    = (unsigned short*)take(5 * wsz * 2);
  unsigned short* qkv   = (unsigned short*)take(5 * (size_t)BHTD * 2);
  unsigned short* attnb = (unsigned short*)take((size_t)BT * Cc * 2);
  unsigned short* pwb   = (unsigned short*)take((size_t)Cc * Cc * 2);
  float* lamp           = (float*)take(256);

  const int thr = 256;
  int nx = BT * Cc;
  cvt_bf16_kernel<<<(nx + thr - 1) / thr, thr, 0, stream>>>(x, xb, nx);
  // w5 slot order: 0=query1, 1=key1, 2=query2, 3=key2, 4=value  -> qkv order q1,k1,q2,k2,v
  cvt_bf16_kernel<<<(int)((wsz + thr - 1) / thr), thr, 0, stream>>>(query1, w5 + 0 * wsz, (int)wsz);
  cvt_bf16_kernel<<<(int)((wsz + thr - 1) / thr), thr, 0, stream>>>(key1,   w5 + 1 * wsz, (int)wsz);
  cvt_bf16_kernel<<<(int)((wsz + thr - 1) / thr), thr, 0, stream>>>(query2, w5 + 2 * wsz, (int)wsz);
  cvt_bf16_kernel<<<(int)((wsz + thr - 1) / thr), thr, 0, stream>>>(key2,   w5 + 3 * wsz, (int)wsz);
  cvt_bf16_kernel<<<(int)((wsz + thr - 1) / thr), thr, 0, stream>>>(value,  w5 + 4 * wsz, (int)wsz);
  int npw = Cc * Cc;
  cvt_bf16_kernel<<<(npw + thr - 1) / thr, thr, 0, stream>>>(proj_w, pwb, npw);
  lam_kernel<<<1, 1, 0, stream>>>(lambdas, lamp);

  proj_gemm_kernel<<<dim3(Tc / 256, 5 * Bc * Hc), 256, 0, stream>>>(xb, w5, qkv);
  diff_attn_kernel<<<dim3(Tc / 64, Bc * Hc), 256, 0, stream>>>(qkv, lamp, attnb);
  out_gemm_kernel<<<dim3(BT / 256, Cc / 64), 256, 0, stream>>>(attnb, pwb, proj_b, out);
}
